// MEWISPool_19413252178643
// MI455X (gfx1250) — matrix operand order repaired and checked by reading, not compiled, not run on hardware
//
#include <hip/hip_runtime.h>
#include <math.h>

#define NN 2048
#define DD 512
#define HH 512
#define GG 8
#define PN 256

typedef __attribute__((ext_vector_type(16))) _Float16 v16h;
typedef __attribute__((ext_vector_type(8)))  _Float16 v8h;
typedef __attribute__((ext_vector_type(8)))  float    v8f;

// ---------------------------------------------------------------- utilities
__global__ void k_zero(float* p, long cnt) {
  long i = (long)blockIdx.x * 256 + threadIdx.x;
  if (i < cnt) p[i] = 0.f;
}

__global__ void k_scatter(float* A, const int* ei, int e2) {
  int i = blockIdx.x * 256 + threadIdx.x;
  if (i >= e2) return;
  int a = ei[i], b = ei[e2 + i];
  if (a != b) A[(long)a * NN + b] = 1.0f;
}

// out[r] = sum_j M[r,j]*v[j]  (+ addv[r]);  v==null -> row sum (degree)
__global__ void k_rowdot(float* out, const float* M, const float* v,
                         const float* addv, int n) {
  __shared__ float red[256];
  int r = blockIdx.x, t = threadIdx.x;
  const float* row = M + (long)r * n;
  float s = 0.f;
  for (int j = t; j < n; j += 256) s += row[j] * (v ? v[j] : 1.0f);
  red[t] = s; __syncthreads();
  for (int o = 128; o > 0; o >>= 1) { if (t < o) red[t] += red[t + o]; __syncthreads(); }
  if (t == 0) out[r] = red[0] + (addv ? addv[r] : 0.f);
}

__global__ void k_cvt16(_Float16* dst, const float* src, long cnt) {
  long i = (long)blockIdx.x * 256 + threadIdx.x;
  if (i < cnt) dst[i] = (_Float16)src[i];
}

// dst[c*rows + r] = (f16) src[r*cols + c]   (optionally squared)
__global__ void k_cvtT16(_Float16* dst, const float* src, int rows, int cols, int square) {
  long i = (long)blockIdx.x * 256 + threadIdx.x;
  if (i >= (long)rows * cols) return;
  int r = (int)(i / cols), c = (int)(i % cols);
  float v = src[i];
  if (square) v *= v;
  dst[(long)c * rows + r] = (_Float16)v;
}

// ---------------------------------------------------------------- WMMA GEMM
// C[M x Nc] = A16[M x K] @ Bt16^T   (Bt16 stored row-major as [Nc x K])
// One wave computes a 32x32 tile (2x2 accumulators): 4 back-to-back
// independent v_wmma per K-step, 2x fragment reuse in both operands.
// MODE 0: store      1: +bias,relu   2: +bias,sigmoid
// MODE 3: pooled-adjacency epilogue: clip(e0+acc,0,1)*(m!=n)*e1[m]*e1[n]
// MODE 4: residual: acc + e0[m*Nc+n]
template <int MODE>
__global__ void k_gemm(float* __restrict__ C,
                       const _Float16* __restrict__ A16,
                       const _Float16* __restrict__ Bt16,
                       const float* __restrict__ e0,
                       const float* __restrict__ e1,
                       int M, int Nc, int K) {
  int lane = threadIdx.x & 31;
  int wave = threadIdx.x >> 5;
  int tilesN = Nc >> 5;                        // 32-wide tiles
  long gid = (long)blockIdx.x * 8 + wave;
  long tot = (long)(M >> 5) * tilesN;
  if (gid >= tot) return;                      // wave-uniform: EXEC stays all-1s
  int tm = (int)(gid / tilesN) << 5;
  int tn = (int)(gid % tilesN) << 5;
  int r  = lane & 15;
  int hi = lane >> 4;

  // A fragment: lane holds row; K slots {hi*8 .. +7} and {hi*8+16 .. +23}
  const _Float16* pa0 = A16 + (long)(tm + r) * K + hi * 8;
  const _Float16* pa1 = pa0 + (long)16 * K;
  // B fragment: lane holds col; K slots {hi*8 .. +7} and {hi*8+16 .. +23}
  const _Float16* pb0 = Bt16 + (long)(tn + r) * K + hi * 8;
  const _Float16* pb1 = pb0 + (long)16 * K;

  v8f acc00 = {}, acc01 = {}, acc10 = {}, acc11 = {};
  for (int k0 = 0; k0 < K; k0 += 32) {
    v8h a0l = *(const v8h*)(pa0 + k0);
    v8h a0h = *(const v8h*)(pa0 + k0 + 16);
    v8h a1l = *(const v8h*)(pa1 + k0);
    v8h a1h = *(const v8h*)(pa1 + k0 + 16);
    v16h a0 = __builtin_shufflevector(a0l, a0h, 0,1,2,3,4,5,6,7,8,9,10,11,12,13,14,15);
    v16h a1 = __builtin_shufflevector(a1l, a1h, 0,1,2,3,4,5,6,7,8,9,10,11,12,13,14,15);
    v16h b0 = __builtin_shufflevector(*(const v8h*)(pb0 + k0), *(const v8h*)(pb0 + k0 + 16), 0,1,2,3,4,5,6,7,8,9,10,11,12,13,14,15);
    v16h b1 = __builtin_shufflevector(*(const v8h*)(pb1 + k0), *(const v8h*)(pb1 + k0 + 16), 0,1,2,3,4,5,6,7,8,9,10,11,12,13,14,15);
    if (k0 + 64 <= K) {                        // gfx1250 global_prefetch_b8
      __builtin_prefetch(pa0 + k0 + 32, 0, 1);
      __builtin_prefetch(pa1 + k0 + 32, 0, 1);
      __builtin_prefetch(pb0 + k0 + 32, 0, 1);
      __builtin_prefetch(pb1 + k0 + 32, 0, 1);
    }
    acc00 = __builtin_amdgcn_wmma_f32_16x16x32_f16(false, a0, false, b0, (short)0, acc00, false, false);
    acc01 = __builtin_amdgcn_wmma_f32_16x16x32_f16(false, a0, false, b1, (short)0, acc01, false, false);
    acc10 = __builtin_amdgcn_wmma_f32_16x16x32_f16(false, a1, false, b0, (short)0, acc10, false, false);
    acc11 = __builtin_amdgcn_wmma_f32_16x16x32_f16(false, a1, false, b1, (short)0, acc11, false, false);
  }

  auto store_tile = [&](const v8f& acc, int mo, int no) {
    int n = tn + no + r;                       // C layout: lane=N, vgpr i -> M
#pragma unroll
    for (int i = 0; i < 8; ++i) {
      int m = tm + mo + hi * 8 + i;
      long idx = (long)m * Nc + n;
      float v = acc[i];
      if (MODE == 0) {
        C[idx] = v;
      } else if (MODE == 1) {
        v += e0[n]; C[idx] = v > 0.f ? v : 0.f;
      } else if (MODE == 2) {
        v += e0[n]; C[idx] = 1.f / (1.f + __expf(-v));
      } else if (MODE == 3) {
        float t = e0[idx] + v;                 // A2 + A3
        t = fminf(fmaxf(t, 0.f), 1.f);
        if (m == n) t = 0.f;
        C[idx] = t * e1[m] * e1[n];
      } else if (MODE == 4) {
        C[idx] = v + e0[idx];
      }
    }
  };
  store_tile(acc00, 0, 0);
  store_tile(acc01, 0, 16);
  store_tile(acc10, 16, 0);
  store_tile(acc11, 16, 16);
}

// ---------------------------------------------------------------- node math
// V = deg*x^2 - 2*x*(A@x) + A@(x*x);  Vn[r] = ||V_r||
__global__ void k_vnorm(float* Vn, const float* x, const float* Ax,
                        const float* Axx, const float* deg) {
  __shared__ float red[256];
  int r = blockIdx.x, t = threadIdx.x;
  float d = deg[r], s = 0.f;
  for (int j = t; j < DD; j += 256) {
    long idx = (long)r * DD + j;
    float xv = x[idx];
    float v = d * xv * xv - 2.f * xv * Ax[idx] + Axx[idx];
    s += v * v;
  }
  red[t] = s; __syncthreads();
  for (int o = 128; o > 0; o >>= 1) { if (t < o) red[t] += red[t + o]; __syncthreads(); }
  if (t == 0) Vn[r] = sqrtf(red[0]);
}

// per-graph softmax entropy (segments are contiguous blocks of 256 nodes)
__global__ void k_entropy(float* ent, float* gparts, const float* Vn) {
  __shared__ float red[256];
  int g = blockIdx.x, t = threadIdx.x;
  float v = Vn[g * PN + t];
  red[t] = v; __syncthreads();
  for (int o = 128; o > 0; o >>= 1) { if (t < o) red[t] = fmaxf(red[t], red[t + o]); __syncthreads(); }
  float m = red[0]; __syncthreads();
  float e = __expf(v - m);
  red[t] = e; __syncthreads();
  for (int o = 128; o > 0; o >>= 1) { if (t < o) red[t] += red[t + o]; __syncthreads(); }
  float p = e / red[0];
  float en = (p > 0.f) ? (-p * __logf(p)) : 0.f;
  ent[g * PN + t] = en;
  __syncthreads();
  red[t] = en; __syncthreads();
  for (int o = 128; o > 0; o >>= 1) { if (t < o) red[t] += red[t + o]; __syncthreads(); }
  if (t == 0) gparts[g] = red[0];
}

// MLP1 first layer: input dim 1 -> outer product
__global__ void k_outer1(float* h, const float* ent2, const float* w, const float* b) {
  long i = (long)blockIdx.x * 256 + threadIdx.x;
  if (i >= (long)NN * HH) return;
  int r = (int)(i / HH), c = (int)(i % HH);
  float v = ent2[r] * w[c] + b[c];
  h[i] = v > 0.f ? v : 0.f;
}

// final 512->1 head + outer sigmoid:  prob = sigmoid(sigmoid(h@w33+b33))
__global__ void k_head(float* prob, const float* h, const float* w, const float* b) {
  __shared__ float red[256];
  int r = blockIdx.x, t = threadIdx.x;
  float s = 0.f;
  for (int j = t; j < HH; j += 256) s += h[(long)r * HH + j] * w[j];
  red[t] = s; __syncthreads();
  for (int o = 128; o > 0; o >>= 1) { if (t < o) red[t] += red[t + o]; __syncthreads(); }
  if (t == 0) {
    float hv = 1.f / (1.f + __expf(-(red[0] + b[0])));
    prob[r] = 1.f / (1.f + __expf(-hv));
  }
}

__global__ void k_dots(float* eparts, float* pparts, const float* entf,
                       const float* prob, const float* Ap) {
  __shared__ float r1[256], r2[256];
  int t = threadIdx.x;
  long i = (long)blockIdx.x * 256 + t;
  float a = 0.f, b = 0.f;
  if (i < NN) { a = entf[i] * prob[i]; b = prob[i] * Ap[i]; }
  r1[t] = a; r2[t] = b; __syncthreads();
  for (int o = 128; o > 0; o >>= 1) {
    if (t < o) { r1[t] += r1[t + o]; r2[t] += r2[t + o]; }
    __syncthreads();
  }
  if (t == 0) { eparts[blockIdx.x] = r1[0]; pparts[blockIdx.x] = r2[0]; }
}

__global__ void k_finalize(float* scal, const float* gparts,
                           const float* eparts, const float* pparts, int nparts) {
  if (threadIdx.x == 0 && blockIdx.x == 0) {
    float g = 0.f, e = 0.f, p = 0.f;
    for (int i = 0; i < GG; ++i) g += gparts[i];
    for (int i = 0; i < nparts; ++i) { e += eparts[i]; p += pparts[i]; }
    scal[0] = g; scal[1] = e; scal[2] = p;
  }
}

// deterministic compacted edge list: prefix over exact integer degrees
__global__ void k_prefix(int* offs, const float* deg) {
  if (threadIdx.x == 0 && blockIdx.x == 0) {
    int a = 0;
    for (int r = 0; r < NN; ++r) { offs[r] = a; a += (int)(deg[r] + 0.5f); }
    offs[NN] = a;
  }
}

__global__ void k_fill(int* li, int* lj, const int* offs, const float* A) {
  int r = blockIdx.x * 256 + threadIdx.x;
  if (r >= NN) return;
  int p = offs[r];
  const float* row = A + (long)r * NN;
  for (int c = 0; c < NN; ++c)
    if (row[c] > 0.f) { li[p] = r; lj[p] = c; ++p; }
}

// bitonic argsort, descending, n=2048, 1024 threads (one comparator each)
__global__ void k_sort(const float* prob, int* order) {
  __shared__ float key[NN];
  __shared__ int   idx[NN];
  int t = threadIdx.x;
  for (int i = t; i < NN; i += 1024) { key[i] = prob[i]; idx[i] = i; }
  __syncthreads();
  for (int len = 2; len <= NN; len <<= 1) {
    for (int str = len >> 1; str > 0; str >>= 1) {
      int pos = 2 * t - (t & (str - 1));
      int a = pos, b = pos + str;
      bool up = (pos & len) == 0;               // descending overall
      float ka = key[a], kb = key[b];
      bool sw = up ? (ka < kb) : (ka > kb);
      if (sw) {
        key[a] = kb; key[b] = ka;
        int ia = idx[a]; idx[a] = idx[b]; idx[b] = ia;
      }
      __syncthreads();
    }
  }
  for (int i = t; i < NN; i += 1024) order[i] = idx[i];
}

// sequential greedy MIS scan: one persistent block, state in LDS.
// The next candidate's adjacency row is double-buffered into LDS with
// ASYNCcnt-tracked global_load_async_to_lds_b128 while the current step
// computes, hiding the only serial-chain memory latency.
__global__ void k_greedy(const float* A, const float* deg, const float* entf,
                         const float* prob, const int* order,
                         const int* li, const int* lj, const int* offs,
                         const float* scal, float* selm, float* sel_out,
                         float* loss_out) {
  __shared__ float rowbuf[2][NN];
  __shared__ float dummy[NN];
  __shared__ float s[NN];
  __shared__ unsigned char sel[NN], rej[NN];
  __shared__ float red[256];
  __shared__ float take_sh;
  int t = threadIdx.x;
  int ne = offs[NN];
  float gamma = scal[0];
  float loss = scal[0] - scal[1] + scal[2];
  for (int i = t; i < NN; i += 256) { dummy[i] = prob[i]; sel[i] = 0; rej[i] = 0; }

  auto issue_row = [&](int v, int buf) {
    const float* src = A + (long)v * NN;
#pragma unroll
    for (int j = 0; j < 2; ++j) {
      int fi = (t + j * 256) * 4;               // 16 B per thread-chunk
      unsigned lds = (unsigned)(uintptr_t)&rowbuf[buf][fi];
      const float* g = src + fi;
      asm volatile("global_load_async_to_lds_b128 %0, %1, off"
                   :: "v"(lds), "v"(g) : "memory");
    }
  };
  issue_row(order[0], 0);                       // ASYNCcnt = 2
  __syncthreads();

  for (int step = 0; step < NN; ++step) {
    int v = order[step];
    int buf = step & 1;
    if (step + 1 < NN) {
      issue_row(order[step + 1], buf ^ 1);      // prefetch next row
      asm volatile("s_wait_asynccnt 0x2" ::: "memory");  // current row landed
    } else {
      asm volatile("s_wait_asynccnt 0x0" ::: "memory");
    }
    __syncthreads();
    const float* arow = rowbuf[buf];
    // build s = dummy*(1-nb), s[v]=1; partial entf@s
    float es = 0.f;
    for (int i = t; i < NN; i += 256) {
      float sv = (arow[i] > 0.f) ? 0.f : dummy[i];
      if (i == v) sv = 1.f;
      s[i] = sv;
      es += entf[i] * sv;
    }
    red[t] = es; __syncthreads();
    for (int o = 128; o > 0; o >>= 1) { if (t < o) red[t] += red[t + o]; __syncthreads(); }
    float entdot = red[0]; __syncthreads();
    // quadratic form s @ (A @ s) over compacted directed edge list
    float q = 0.f;
    for (int e = t; e < ne; e += 256) q += s[li[e]] * s[lj[e]];
    red[t] = q; __syncthreads();
    for (int o = 128; o > 0; o >>= 1) { if (t < o) red[t] += red[t + o]; __syncthreads(); }
    float quad = red[0]; __syncthreads();
    if (t == 0) {
      float l = gamma - entdot + quad;
      bool has_nb = deg[v] > 0.f;
      bool elig = !(rej[v] | sel[v]);
      bool take = has_nb && elig && (l <= loss);
      if (!has_nb || take) sel[v] = 1;
      take_sh = take ? 1.f : 0.f;
    }
    __syncthreads();
    if (take_sh > 0.f) {
      for (int i = t; i < NN; i += 256) {
        dummy[i] = s[i];
        if (arow[i] > 0.f) rej[i] = 1;
      }
    }
    __syncthreads();
  }
  for (int i = t; i < NN; i += 256) {
    float f = sel[i] ? 1.f : 0.f;
    selm[i] = f; sel_out[i] = f;
  }
  if (t == 0) loss_out[0] = loss;
}

__global__ void k_xpool(float* out, const float* x, const float* selm) {
  long i = (long)blockIdx.x * 256 + threadIdx.x;
  if (i >= (long)NN * DD) return;
  out[i] = x[i] * selm[i / DD];
}

// ---------------------------------------------------------------- launcher
static inline int cdiv(long a, long b) { return (int)((a + b - 1) / b); }

extern "C" void kernel_launch(void* const* d_in, const int* in_sizes, int n_in,
                              void* d_out, int out_size, void* d_ws, size_t ws_size,
                              hipStream_t stream) {
  const float* x    = (const float*)d_in[0];
  const int*   ei   = (const int*)d_in[1];
  // d_in[2] = batch (unused; segments are contiguous 256-node blocks)
  const float* w11 = (const float*)d_in[3],  *b11 = (const float*)d_in[4];
  const float* w12 = (const float*)d_in[5],  *b12 = (const float*)d_in[6];
  const float* w13 = (const float*)d_in[7],  *b13 = (const float*)d_in[8];
  const float* w21 = (const float*)d_in[9],  *b21 = (const float*)d_in[10];
  const float* w22 = (const float*)d_in[11], *b22 = (const float*)d_in[12];
  const float* w23 = (const float*)d_in[13], *b23 = (const float*)d_in[14];
  const float* w31 = (const float*)d_in[15], *b31 = (const float*)d_in[16];
  const float* w32 = (const float*)d_in[17], *b32 = (const float*)d_in[18];
  const float* w33 = (const float*)d_in[19], *b33 = (const float*)d_in[20];
  int e2 = in_sizes[1] / 2;                  // 32768 directed edge entries

  // ---- workspace layout (~65 MiB)
  char* base = (char*)d_ws;
  size_t cur = 0;
  auto alloc = [&](size_t bytes) -> void* {
    void* p = base + cur;
    cur = (cur + bytes + 255) & ~(size_t)255;
    return p;
  };
  float*     A     = (float*)    alloc((size_t)NN * NN * 4);
  _Float16*  A16   = (_Float16*) alloc((size_t)NN * NN * 2);
  _Float16*  xT16  = (_Float16*) alloc((size_t)DD * NN * 2);
  _Float16*  xxT16 = (_Float16*) alloc((size_t)DD * NN * 2);
  float*     hA    = (float*)    alloc((size_t)NN * HH * 4);   // also Ax
  float*     hB    = (float*)    alloc((size_t)NN * HH * 4);   // also A@(x*x)
  _Float16*  h16   = (_Float16*) alloc((size_t)NN * HH * 2);
  _Float16*  hT16  = (_Float16*) alloc((size_t)HH * NN * 2);
  _Float16*  wT16  = (_Float16*) alloc((size_t)HH * HH * 2);
  float*     A2    = (float*)    alloc((size_t)NN * NN * 4);
  _Float16*  A216  = (_Float16*) alloc((size_t)NN * NN * 2);
  float*     deg   = (float*)    alloc(NN * 4);
  float*     Vn    = (float*)    alloc(NN * 4);
  float*     ent   = (float*)    alloc(NN * 4);
  float*     ent2  = (float*)    alloc(NN * 4);
  float*     prob  = (float*)    alloc(NN * 4);
  float*     Ap    = (float*)    alloc(NN * 4);
  float*     selm  = (float*)    alloc(NN * 4);
  int*       order = (int*)      alloc(NN * 4);
  int*       offs  = (int*)      alloc((NN + 1) * 4);
  int*       eli   = (int*)      alloc(65536 * 4);
  int*       elj   = (int*)      alloc(65536 * 4);
  float*     scal  = (float*)    alloc(16 * 4);
  float*     gparts= (float*)    alloc(GG * 4);
  float*     eparts= (float*)    alloc(8 * 4);
  float*     pparts= (float*)    alloc(8 * 4);

  float* out  = (float*)d_out;
  float* oxp  = out;                          // x_pooled   [NN*DD]
  float* oadj = out + (long)NN * DD;          // adj_pooled [NN*NN]
  float* osel = oadj + (long)NN * NN;         // sel        [NN]
  float* olos = osel + NN;                    // loss       [1]

  auto gemm = [&](int mode, float* C, const _Float16* Aop, const _Float16* Bt,
                  const float* e0, const float* e1, int M, int Nc, int K) {
    int blocks = (M / 32) * (Nc / 32) / 8;    // 32x32 tile per wave, 8 waves/block
    switch (mode) {
      case 0: k_gemm<0><<<blocks, 256, 0, stream>>>(C, Aop, Bt, e0, e1, M, Nc, K); break;
      case 1: k_gemm<1><<<blocks, 256, 0, stream>>>(C, Aop, Bt, e0, e1, M, Nc, K); break;
      case 2: k_gemm<2><<<blocks, 256, 0, stream>>>(C, Aop, Bt, e0, e1, M, Nc, K); break;
      case 3: k_gemm<3><<<blocks, 256, 0, stream>>>(C, Aop, Bt, e0, e1, M, Nc, K); break;
      case 4: k_gemm<4><<<blocks, 256, 0, stream>>>(C, Aop, Bt, e0, e1, M, Nc, K); break;
    }
  };

  // 1) adjacency
  k_zero<<<cdiv((long)NN * NN, 256), 256, 0, stream>>>(A, (long)NN * NN);
  k_scatter<<<cdiv(e2, 256), 256, 0, stream>>>(A, ei, e2);
  k_rowdot<<<NN, 256, 0, stream>>>(deg, A, nullptr, nullptr, NN);
  k_cvt16<<<cdiv((long)NN * NN, 256), 256, 0, stream>>>(A16, A, (long)NN * NN);

  // 2) V = deg*x^2 - 2*x*(A@x) + A@(x*x), row norms, entropy
  k_cvtT16<<<cdiv((long)NN * DD, 256), 256, 0, stream>>>(xT16,  x, NN, DD, 0);
  k_cvtT16<<<cdiv((long)NN * DD, 256), 256, 0, stream>>>(xxT16, x, NN, DD, 1);
  gemm(0, hA, A16, xT16,  nullptr, nullptr, NN, DD, NN);   // A@x
  gemm(0, hB, A16, xxT16, nullptr, nullptr, NN, DD, NN);   // A@(x*x)
  k_vnorm<<<NN, 256, 0, stream>>>(Vn, x, hA, hB, deg);
  k_entropy<<<GG, PN, 0, stream>>>(ent, gparts, Vn);
  k_rowdot<<<NN, 256, 0, stream>>>(ent2, A, ent, ent, NN); // ent + A@ent

  // 3) MLP1
  k_outer1<<<cdiv((long)NN * HH, 256), 256, 0, stream>>>(hA, ent2, w11, b11);
  k_cvt16<<<cdiv((long)NN * HH, 256), 256, 0, stream>>>(h16, hA, (long)NN * HH);
  k_cvtT16<<<cdiv((long)HH * HH, 256), 256, 0, stream>>>(wT16, w12, HH, HH, 0);
  gemm(1, hB, h16, wT16, b12, nullptr, NN, HH, HH);
  k_cvt16<<<cdiv((long)NN * HH, 256), 256, 0, stream>>>(h16, hB, (long)NN * HH);
  k_cvtT16<<<cdiv((long)HH * HH, 256), 256, 0, stream>>>(wT16, w13, HH, HH, 0);
  gemm(2, hA, h16, wT16, b13, nullptr, NN, HH, HH);        // h1out = hA

  // 4) hin2 = h1 + A@h1 ; MLP2
  k_cvtT16<<<cdiv((long)NN * HH, 256), 256, 0, stream>>>(hT16, hA, NN, HH, 0);
  gemm(4, hB, A16, hT16, hA, nullptr, NN, HH, NN);
  k_cvt16<<<cdiv((long)NN * HH, 256), 256, 0, stream>>>(h16, hB, (long)NN * HH);
  k_cvtT16<<<cdiv((long)HH * HH, 256), 256, 0, stream>>>(wT16, w21, HH, HH, 0);
  gemm(1, hA, h16, wT16, b21, nullptr, NN, HH, HH);
  k_cvt16<<<cdiv((long)NN * HH, 256), 256, 0, stream>>>(h16, hA, (long)NN * HH);
  k_cvtT16<<<cdiv((long)HH * HH, 256), 256, 0, stream>>>(wT16, w22, HH, HH, 0);
  gemm(1, hB, h16, wT16, b22, nullptr, NN, HH, HH);
  k_cvt16<<<cdiv((long)NN * HH, 256), 256, 0, stream>>>(h16, hB, (long)NN * HH);
  k_cvtT16<<<cdiv((long)HH * HH, 256), 256, 0, stream>>>(wT16, w23, HH, HH, 0);
  gemm(2, hA, h16, wT16, b23, nullptr, NN, HH, HH);        // h2out = hA

  // 5) hin3 = h2 + A@h2 ; MLP3
  k_cvtT16<<<cdiv((long)NN * HH, 256), 256, 0, stream>>>(hT16, hA, NN, HH, 0);
  gemm(4, hB, A16, hT16, hA, nullptr, NN, HH, NN);
  k_cvt16<<<cdiv((long)NN * HH, 256), 256, 0, stream>>>(h16, hB, (long)NN * HH);
  k_cvtT16<<<cdiv((long)HH * HH, 256), 256, 0, stream>>>(wT16, w31, HH, HH, 0);
  gemm(1, hA, h16, wT16, b31, nullptr, NN, HH, HH);
  k_cvt16<<<cdiv((long)NN * HH, 256), 256, 0, stream>>>(h16, hA, (long)NN * HH);
  k_cvtT16<<<cdiv((long)HH * HH, 256), 256, 0, stream>>>(wT16, w32, HH, HH, 0);
  gemm(1, hB, h16, wT16, b32, nullptr, NN, HH, HH);
  k_head<<<NN, 256, 0, stream>>>(prob, hB, w33, b33);

  // 6) loss pieces + deterministic edge compaction + argsort + greedy scan
  k_rowdot<<<NN, 256, 0, stream>>>(Ap, A, prob, nullptr, NN);
  k_dots<<<NN / 256, 256, 0, stream>>>(eparts, pparts, ent, prob, Ap);
  k_finalize<<<1, 32, 0, stream>>>(scal, gparts, eparts, pparts, NN / 256);
  k_prefix<<<1, 32, 0, stream>>>(offs, deg);
  k_fill<<<NN / 256, 256, 0, stream>>>(eli, elj, offs, A);
  k_sort<<<1, 1024, 0, stream>>>(prob, order);
  k_greedy<<<1, 256, 0, stream>>>(A, deg, ent, prob, order, eli, elj, offs,
                                  scal, selm, osel, olos);

  // 7) outputs: x_pooled, A2 = A@A, adj_pooled = clip(A2+A2@A,0,1)*(1-I)*s s^T
  k_xpool<<<cdiv((long)NN * DD, 256), 256, 0, stream>>>(oxp, x, selm);
  gemm(0, A2, A16, A16, nullptr, nullptr, NN, NN, NN);     // A symmetric: A16 is its own Bt
  k_cvt16<<<cdiv((long)NN * NN, 256), 256, 0, stream>>>(A216, A2, (long)NN * NN);
  gemm(3, oadj, A216, A16, A2, selm, NN, NN, NN);
  (void)n_in; (void)out_size; (void)ws_size;
}
